// TransImg2_23261542875589
// MI455X (gfx1250) — matrix operand order repaired and checked by reading, not compiled, not run on hardware
//
#include <hip/hip_runtime.h>
#include <hip/hip_bf16.h>
#include <math.h>

// ---------------------------------------------------------------------------
// Types for CDNA5 WMMA
// ---------------------------------------------------------------------------
typedef __attribute__((ext_vector_type(16))) __bf16 v16bf;
typedef __attribute__((ext_vector_type(8)))  __bf16 v8bf;
typedef __attribute__((ext_vector_type(8)))  float  v8f;

struct BfPair { v8bf a, b; };   // 32 bytes, bit_cast-able to v16bf

static constexpr int N_NODES = 10000;
static constexpr int N_PAD   = 10048;   // 157 * 64, removes all row guards
static constexpr int E_EDGES = 160000;

__device__ __forceinline__ unsigned short f2bf_bits(float f) {
    unsigned u = __float_as_uint(f);
    return (unsigned short)((u + 0x7FFFu + ((u >> 16) & 1u)) >> 16);
}
__device__ __forceinline__ __bf16 f2bf(float f) {
    unsigned short s = f2bf_bits(f);
    return __builtin_bit_cast(__bf16, s);
}
__device__ __forceinline__ unsigned pk2bf(float lo, float hi) {
    return ((unsigned)f2bf_bits(hi) << 16) | (unsigned)f2bf_bits(lo);
}

__device__ __forceinline__ float atomicMaxF(float* addr, float val) {
    if (val >= 0.0f)
        return __int_as_float(atomicMax((int*)addr, __float_as_int(val)));
    else
        return __uint_as_float(atomicMin((unsigned int*)addr, __float_as_uint(val)));
}

template <int X> struct Log2      { static constexpr int v = 1 + Log2<X / 2>::v; };
template <>      struct Log2<1>   { static constexpr int v = 0; };

// ---------------------------------------------------------------------------
// Per-layer weight prep: concat Wq|Wk|Wv|Ws -> bf16 [din, 4*dout], biases [4*dout]
// ---------------------------------------------------------------------------
template <int DIN, int DOUT>
__global__ void prep_layer(const float* __restrict__ Wq, const float* __restrict__ bq,
                           const float* __restrict__ Wk, const float* __restrict__ bk,
                           const float* __restrict__ Wv, const float* __restrict__ bv,
                           const float* __restrict__ Ws, const float* __restrict__ bs,
                           __bf16* __restrict__ wbf, float* __restrict__ bcat) {
    constexpr int DOUT4 = DOUT * 4;
    constexpr int LB    = Log2<DOUT>::v;
    const int total = DIN * DOUT4;
    for (int i = blockIdx.x * blockDim.x + threadIdx.x; i < total;
         i += gridDim.x * blockDim.x) {
        int k = i >> (LB + 2);
        int c = i & (DOUT4 - 1);
        int g = c >> LB;
        int n = c & (DOUT - 1);
        const float* W = (g == 0) ? Wq : (g == 1) ? Wk : (g == 2) ? Wv : Ws;
        wbf[i] = f2bf(W[k * DOUT + n]);
    }
    for (int c = blockIdx.x * blockDim.x + threadIdx.x; c < DOUT4;
         c += gridDim.x * blockDim.x) {
        int g = c >> LB;
        int n = c & (DOUT - 1);
        const float* B = (g == 0) ? bq : (g == 1) ? bk : (g == 2) ? bv : bs;
        bcat[c] = B[n];
    }
}

// ---------------------------------------------------------------------------
// Fused Q|K|V|Skip GEMM:  C[N, 4*DOUT] = X[N,DIN] @ Wbf[DIN,4*DOUT] + bias
// Block tile 64(M) x 64(N), 8 waves (4M x 2N), each wave: one A fragment reused
// by two v_wmma_f32_16x16x32_bf16 per K-step. Branch-free main loop:
// A loads use clamped row index; outputs are padded to N_PAD rows.
// LDS: A row-major [64][32] (per-lane 16B chunks), B column-major [64][32]
// (per-lane fragments fully contiguous -> ds_load_b128 pairs, no repacking).
// ---------------------------------------------------------------------------
template <int DIN, int DOUT>
__global__ __launch_bounds__(256)
void gemm_qkvs(const float* __restrict__ X, const __bf16* __restrict__ Wbf,
               const float* __restrict__ bcat,
               float* __restrict__ Q, float* __restrict__ Kq,
               float* __restrict__ V, float* __restrict__ O) {
    constexpr int DOUT4 = DOUT * 4;
    constexpr int LB    = Log2<DOUT>::v;
    __shared__ __bf16 As[64 * 32];      // [row][k], stride 32 halves
    __shared__ __bf16 Bs[64 * 32];      // [col][k] column-major, stride 32 halves

    const int tid   = threadIdx.x;
    const int lane  = tid & 31;
    const int wid   = tid >> 5;
    const int waveM = wid & 3;          // 4 M-tiles of 16
    const int waveN = wid >> 2;         // 2 N-slabs of 32
    const int rowBase = blockIdx.y * 64;
    const int colBase = blockIdx.x * 64;
    const int hi = lane >> 4;
    const int lo = lane & 15;

    // A staging map: thread -> (row r = tid>>2, cols c0 = (tid&3)*8 .. +7)
    const int ar  = tid >> 2;
    const int ac0 = (tid & 3) * 8;
    const int arow  = rowBase + ar;
    const int arowc = (arow < N_NODES) ? arow : (N_NODES - 1);   // clamp, branchless
    // B staging map: thread -> (k-row br = tid>>3, col group bc0 = (tid&7)*8)
    const int br  = tid >> 3;
    const int bc0 = (tid & 7) * 8;

    v8f acc0 = {};
    v8f acc1 = {};

    for (int k0 = 0; k0 < DIN; k0 += 32) {
        // ---- Stage A: 2 x float4 load, pack to bf16, one 16B LDS store ----
        {
            const float* xp = X + (size_t)arowc * DIN + k0 + ac0;
            float4 x0 = *(const float4*)(xp);
            float4 x1 = *(const float4*)(xp + 4);
            uint4 pk;
            pk.x = pk2bf(x0.x, x0.y);
            pk.y = pk2bf(x0.z, x0.w);
            pk.z = pk2bf(x1.x, x1.y);
            pk.w = pk2bf(x1.z, x1.w);
            *(uint4*)(As + ar * 32 + ac0) = pk;
            if (k0 + 32 < DIN)
                __builtin_prefetch(xp + 32, 0, 1);
        }
        // ---- Stage B: 16B global read of bf16, transposed scalar LDS writes ----
        {
            uint4 w = *(const uint4*)(Wbf + (size_t)(k0 + br) * DOUT4 + colBase + bc0);
            union { uint4 q; __bf16 h[8]; } u;
            u.q = w;
            #pragma unroll
            for (int j = 0; j < 8; ++j)
                Bs[(bc0 + j) * 32 + br] = u.h[j];
        }
        __syncthreads();

        // ---- Fragments (vector LDS loads, no repacking) ----
        const int am = waveM * 16 + lo;
        v16bf af = *(const v16bf*)(As + am * 32 + 16 * hi);

        const int bn = waveN * 32 + lo;
        v16bf bf0 = *(const v16bf*)(Bs + bn * 32 + 16 * hi);
        v16bf bf1 = *(const v16bf*)(Bs + (bn + 16) * 32 + 16 * hi);

        acc0 = __builtin_amdgcn_wmma_f32_16x16x32_bf16(
                 false, af, false, bf0, (short)0, acc0, false, false);
        acc1 = __builtin_amdgcn_wmma_f32_16x16x32_bf16(
                 false, af, false, bf1, (short)0, acc1, false, false);
        __syncthreads();
    }

    // C/D layout: VGPR r -> M = r + 8*(lane>>4), N = lane&15. No row guards (padded).
    #pragma unroll
    for (int t = 0; t < 2; ++t) {
        const int colb = colBase + waveN * 32 + t * 16;   // wave-uniform base
        const int g    = colb >> LB;                      // which of Q/K/V/S
        const int ccb  = colb & (DOUT - 1);
        const float bias = bcat[colb + lo];
        float* outp = (g == 0) ? Q : (g == 1) ? Kq : (g == 2) ? V : O;
        const v8f& acc = t ? acc1 : acc0;
        const int row0 = rowBase + waveM * 16 + hi * 8;
        #pragma unroll
        for (int r = 0; r < 8; ++r) {
            outp[(size_t)(row0 + r) * DOUT + ccb + lo] = acc[r] + bias;
        }
    }
}

// ---------------------------------------------------------------------------
// Edge phase kernels
// ---------------------------------------------------------------------------
__global__ void init_md(float* __restrict__ m, float* __restrict__ denom, int n) {
    int i = blockIdx.x * blockDim.x + threadIdx.x;
    if (i < n) { m[i] = __int_as_float(0xFF800000); denom[i] = 0.0f; }
}

// one wave32 per edge: logits[e] = scale * dot(q[dst], k[src]); segment max via atomics
template <int D>
__global__ __launch_bounds__(256)
void edge_logits(const float* __restrict__ Q, const float* __restrict__ Kf,
                 const int* __restrict__ src, const int* __restrict__ dst,
                 float* __restrict__ logits, float* __restrict__ m,
                 int E_, float scale) {
    int e = blockIdx.x * (blockDim.x >> 5) + (threadIdx.x >> 5);
    if (e >= E_) return;
    int lane = threadIdx.x & 31;
    int s = src[e], t = dst[e];
    const float4* q4 = (const float4*)(Q + (size_t)t * D);
    const float4* k4 = (const float4*)(Kf + (size_t)s * D);
    float acc = 0.0f;
    #pragma unroll
    for (int it = 0; it < D / 128; ++it) {
        int j = lane + it * 32;
        float4 a = q4[j], b = k4[j];
        acc += a.x * b.x + a.y * b.y + a.z * b.z + a.w * b.w;
    }
    #pragma unroll
    for (int off = 16; off > 0; off >>= 1) acc += __shfl_down(acc, off, 32);
    if (lane == 0) {
        float lg = acc * scale;
        logits[e] = lg;
        atomicMaxF(&m[t], lg);
    }
}

__global__ void edge_exp(const float* __restrict__ logits, const float* __restrict__ m,
                         const int* __restrict__ dst,
                         float* __restrict__ expw, float* __restrict__ denom, int E_) {
    int e = blockIdx.x * blockDim.x + threadIdx.x;
    if (e >= E_) return;
    int t = dst[e];
    float w = __expf(logits[e] - m[t]);
    expw[e] = w;
    unsafeAtomicAdd(&denom[t], w);
}

// one wave32 per edge: O[dst] += (expw/denom[dst]) * V[src]  (hardware f32 atomics)
template <int D>
__global__ __launch_bounds__(256)
void edge_agg(const float* __restrict__ V, const float* __restrict__ expw,
              const float* __restrict__ denom,
              const int* __restrict__ src, const int* __restrict__ dst,
              float* __restrict__ O, int E_) {
    int e = blockIdx.x * (blockDim.x >> 5) + (threadIdx.x >> 5);
    if (e >= E_) return;
    int lane = threadIdx.x & 31;
    int s = src[e], t = dst[e];
    float w = expw[e] / denom[t];
    const float4* v4 = (const float4*)(V + (size_t)s * D);
    float* op = O + (size_t)t * D;
    #pragma unroll
    for (int it = 0; it < D / 128; ++it) {
        int j = lane + it * 32;
        float4 a = v4[j];
        int base = j << 2;
        unsafeAtomicAdd(op + base + 0, w * a.x);
        unsafeAtomicAdd(op + base + 1, w * a.y);
        unsafeAtomicAdd(op + base + 2, w * a.z);
        unsafeAtomicAdd(op + base + 3, w * a.w);
    }
}

// ---------------------------------------------------------------------------
// Finalize
// ---------------------------------------------------------------------------
__global__ void elu_k(const float* __restrict__ in, float* __restrict__ out, int n) {
    int i = blockIdx.x * blockDim.x + threadIdx.x;
    if (i < n) {
        float x = in[i];
        out[i] = (x > 0.0f) ? x : (__expf(x) - 1.0f);
    }
}

__global__ void copy2_k(const float* __restrict__ in, float* __restrict__ o1,
                        float* __restrict__ o2, int n) {
    int i = blockIdx.x * blockDim.x + threadIdx.x;
    if (i < n) {
        float x = in[i];
        o1[i] = x;
        if (o2) o2[i] = x;
    }
}

// ---------------------------------------------------------------------------
// Host orchestration
// ---------------------------------------------------------------------------
extern "C" void kernel_launch(void* const* d_in, const int* in_sizes, int n_in,
                              void* d_out, int out_size, void* d_ws, size_t ws_size,
                              hipStream_t stream) {
    (void)in_sizes; (void)n_in; (void)out_size; (void)ws_size;

    const int N = N_NODES, E = E_EDGES;
    // d_in[0] = features (unused by the reference), d_in[1] = img_feat, d_in[2] = edge_index
    const float* img  = (const float*)d_in[1];
    const int*   ei   = (const int*)d_in[2];
    const int*   srcp = ei;          // row 0 = src
    const int*   dstp = ei + E;      // row 1 = dst

    // Workspace layout (f32 unless noted); node buffers padded to N_PAD rows
    float* ws = (float*)d_ws;
    const size_t NC = (size_t)N_PAD * 512;
    float* xbuf   = ws;                 // current layer input  [N_PAD,512]
    float* qbuf   = xbuf + NC;
    float* kbuf   = qbuf + NC;
    float* vbuf   = kbuf + NC;
    float* obuf   = vbuf + NC;          // skip + aggregation   [N_PAD,512]
    float* logits = obuf + NC;          // [E]
    float* expw   = logits + E;         // [E]
    float* mbuf   = expw + E;           // [N]
    float* dbuf   = mbuf + N;           // [N]
    float* bcat   = dbuf + N;           // [2048]
    __bf16* wbf   = (__bf16*)(bcat + 2048);   // [din, 4*dout] bf16, <= 1 MB

    for (int l = 0; l < 4; ++l) {
        const bool wide_in = (l == 1 || l == 3);    // din=512,dout=256 for layers 2&4
        const int dout  = wide_in ? 256 : 512;
        const int dout4 = dout * 4;
        const float* X = (l == 0) ? img : xbuf;

        // params: d_in[3 + 8*l ... +7] = Wq,bq,Wk,bk,Wv,bv,Ws,bs
        const float* P[8];
        for (int i = 0; i < 8; ++i) P[i] = (const float*)d_in[3 + 8 * l + i];

        dim3 ggrid(dout4 / 64, N_PAD / 64);
        if (!wide_in) {
            prep_layer<256, 512><<<1024, 256, 0, stream>>>(
                P[0], P[1], P[2], P[3], P[4], P[5], P[6], P[7], wbf, bcat);
            gemm_qkvs<256, 512><<<ggrid, 256, 0, stream>>>(
                X, wbf, bcat, qbuf, kbuf, vbuf, obuf);
        } else {
            prep_layer<512, 256><<<1024, 256, 0, stream>>>(
                P[0], P[1], P[2], P[3], P[4], P[5], P[6], P[7], wbf, bcat);
            gemm_qkvs<512, 256><<<ggrid, 256, 0, stream>>>(
                X, wbf, bcat, qbuf, kbuf, vbuf, obuf);
        }

        init_md<<<(N + 255) / 256, 256, 0, stream>>>(mbuf, dbuf, N);

        const float scale = 1.0f / sqrtf((float)dout);
        if (!wide_in) {
            edge_logits<512><<<(E + 7) / 8, 256, 0, stream>>>(
                qbuf, kbuf, srcp, dstp, logits, mbuf, E, scale);
        } else {
            edge_logits<256><<<(E + 7) / 8, 256, 0, stream>>>(
                qbuf, kbuf, srcp, dstp, logits, mbuf, E, scale);
        }
        edge_exp<<<(E + 255) / 256, 256, 0, stream>>>(logits, mbuf, dstp, expw, dbuf, E);
        if (!wide_in) {
            edge_agg<512><<<(E + 7) / 8, 256, 0, stream>>>(
                vbuf, expw, dbuf, srcp, dstp, obuf, E);
        } else {
            edge_agg<256><<<(E + 7) / 8, 256, 0, stream>>>(
                vbuf, expw, dbuf, srcp, dstp, obuf, E);
        }

        const int nel = N * dout;
        if (l == 0 || l == 2) {
            elu_k<<<(nel + 255) / 256, 256, 0, stream>>>(obuf, xbuf, nel);
        } else {
            float* oseg = (float*)d_out + ((l == 1) ? 0 : (size_t)N * 256);
            float* next = (l == 1) ? xbuf : nullptr;   // img2 feeds layer 3
            copy2_k<<<(nel + 255) / 256, 256, 0, stream>>>(obuf, oseg, next, nel);
        }
    }
}
